// LightGlueAttention_12146167513248
// MI455X (gfx1250) — matrix operand order repaired and checked by reading, not compiled, not run on hardware
//
#include <hip/hip_runtime.h>
#include <math.h>

// LightGlue eager attention for MI455X (gfx1250), wave32 + WMMA bf16.
// d_out = output[B,H,N,D] (f32) ++ attention[B,H,N,N] (f32).
//
// S^T formulation: S^T = K.Q^T (A = K tile, B = Q^T fixed in registers).
// C-layout puts the query row on the lane index and 8 consecutive keys in the
// accumulator VGPRs: softmax stats are per-lane register reductions (one
// xor-16 shuffle per pass, total) and attention/output stores are contiguous
// b128 per lane. Three passes: (1) row max, (2) row sum, (3) probabilities
// written once + O^T = V^T.P^T. Softmax in the log2 domain with the
// normalizer folded into the exponent: p = exp2(s - (m + log2(l))).
// LDS tiles are double-buffered and software-pipelined; the j-loop is
// unrolled by 2 (compile-time buffer indices -> immediate ds offsets) and the
// last tile-pair is peeled so the steady-state loop is branchless.

typedef __attribute__((ext_vector_type(16))) __bf16 v16bf;
typedef __attribute__((ext_vector_type(8)))  float  v8f;
typedef unsigned int uint32;

#define N_CTX 4096
#define DHEAD 64
#define BH    16                    // B*H
#define WAVES 8
#define ROWS_PER_WG (16 * WAVES)    // 128 query rows per workgroup
#define KTILE 64                    // keys staged in LDS per iteration
#define NJ    (N_CTX / KTILE)       // 64 key tiles (even)
#define KPITCH 66                   // bf16 pitch (33 dwords, coprime w/ 64 banks)
#define PPITCH 18                   // bf16 pitch for P^T staging

// A-operand (16-bit, 16x32) per-lane K index for element e of v16bf.
__device__ __forceinline__ int a_kidx(int e, int half) {
  int j = e >> 1, h = e & 1;
  return ((j & 3) * 2 + h) + ((j >= 4) ? 16 : 0) + (half ? 8 : 0);
}

// one v_cvt_pk_bf16_f32: pack two f32 into a bf16x2 dword (RNE)
__device__ __forceinline__ uint32 pk2(float a, float b) {
  uint32 r;
  asm("v_cvt_pk_bf16_f32 %0, %1, %2" : "=v"(r) : "v"(a), "v"(b));
  return r;
}

__global__ __launch_bounds__(256, 2)
void lg_attn_kernel(const float* __restrict__ q, const float* __restrict__ k,
                    const float* __restrict__ v, float* __restrict__ out,
                    float* __restrict__ attn) {
  __shared__ __bf16 kS[2][KTILE * KPITCH];       // K tile, row-major [key][d]
  __shared__ __bf16 vT[2][DHEAD * KPITCH];       // V tile, transposed [d][key]
  __shared__ __bf16 pS[WAVES][32 * PPITCH];      // per-wave P^T chunk [key][row]

  const int tid  = threadIdx.x;
  const int wave = tid >> 5;
  const int lane = tid & 31;
  const int half = lane >> 4;
  const int l16  = lane & 15;
  const int sky  = tid >> 4;                     // staging: key row
  const int sd   = (tid & 15) * 4;               // staging: d (float4 granularity)

  const int bh   = blockIdx.x >> 5;              // 16 heads
  const int rb   = blockIdx.x & 31;              // 32 row-blocks per head
  const int row0 = rb * ROWS_PER_WG + wave * 16;

  const float* qH = q + (size_t)bh * N_CTX * DHEAD;
  const float* kH = k + (size_t)bh * N_CTX * DHEAD;
  const float* vH = v + (size_t)bh * N_CTX * DHEAD;
  float* outH  = out  + (size_t)bh * N_CTX * DHEAD;
  float* attnH = attn + (size_t)bh * N_CTX * N_CTX;

  // D^-1/2 * log2(e): softmax in the log2 domain.
  const float qscale = 0.125f * 1.44269504088896340736f;

  // ---- Q^T as the fixed B operand: B[k=d][n=row], lane = row, element = d.
  v16bf bq0, bq1;
  #pragma unroll
  for (int e = 0; e < 16; ++e) {
    bq0[e] = (__bf16)(qH[(size_t)(row0 + l16) * DHEAD + a_kidx(e, half)]      * qscale);
    bq1[e] = (__bf16)(qH[(size_t)(row0 + l16) * DHEAD + 32 + a_kidx(e, half)] * qscale);
  }

  // ============ staging helpers (4 float4 per thread per tile) ============
  float4 gk[4], gv[4];
  #define LOAD_K(key0) _Pragma("unroll") \
    for (int i = 0; i < 4; ++i) \
      gk[i] = *(const float4*)&kH[(size_t)((key0) + sky + i * 16) * DHEAD + sd];
  #define LOAD_V(key0) _Pragma("unroll") \
    for (int i = 0; i < 4; ++i) \
      gv[i] = *(const float4*)&vH[(size_t)((key0) + sky + i * 16) * DHEAD + sd];
  #define STORE_K(buf) _Pragma("unroll") \
    for (int i = 0; i < 4; ++i) { \
      uint32* p = (uint32*)&kS[buf][(sky + i * 16) * KPITCH + sd]; \
      p[0] = pk2(gk[i].x, gk[i].y); \
      p[1] = pk2(gk[i].z, gk[i].w); \
    }
  #define STORE_V(buf) _Pragma("unroll") \
    for (int i = 0; i < 4; ++i) { \
      const int ky = sky + i * 16; \
      vT[buf][(sd + 0) * KPITCH + ky] = (__bf16)gv[i].x; \
      vT[buf][(sd + 1) * KPITCH + ky] = (__bf16)gv[i].y; \
      vT[buf][(sd + 2) * KPITCH + ky] = (__bf16)gv[i].z; \
      vT[buf][(sd + 3) * KPITCH + ky] = (__bf16)gv[i].w; \
    }
  #define PREF_K(j) __builtin_prefetch(&kH[(size_t)(j) * KTILE * DHEAD + tid * 16], 0, 3)
  #define PREF_V(j) __builtin_prefetch(&vH[(size_t)(j) * KTILE * DHEAD + tid * 16], 0, 3)

  // S^T subtile: A = K rows from LDS (immediate offsets off a hoisted base),
  // B = Q^T in registers. c += 16 keys x 16 rows.
  #define QK_TILE(c, buf, n) { \
    const __bf16* ap = &kS[buf][((n) * 16 + l16) * KPITCH]; \
    v16bf a0, a1; \
    _Pragma("unroll") \
    for (int e = 0; e < 16; ++e) { \
      const int kk = a_kidx(e, half); \
      a0[e] = ap[kk]; \
      a1[e] = ap[32 + kk]; \
    } \
    c = __builtin_amdgcn_wmma_f32_16x16x32_bf16(false, a0, false, bq0, (short)0, c, false, false); \
    c = __builtin_amdgcn_wmma_f32_16x16x32_bf16(false, a1, false, bq1, (short)0, c, false, false); \
  }

  #define P1_BODY(buf) { \
    _Pragma("unroll") \
    for (int n = 0; n < 4; ++n) { \
      v8f c = {}; \
      QK_TILE(c, buf, n); \
      _Pragma("unroll") \
      for (int r = 0; r < 8; ++r) mrow = fmaxf(mrow, c[r]); \
    } }

  #define P2_BODY(buf) { \
    _Pragma("unroll") \
    for (int n = 0; n < 4; ++n) { \
      v8f c = {}; \
      QK_TILE(c, buf, n); \
      float s = 0.f; \
      _Pragma("unroll") \
      for (int r = 0; r < 8; ++r) s += __builtin_exp2f(c[r] - mrow); \
      lsum += s; \
    } }

  #define P3_BODY(buf, key0) { \
    _Pragma("unroll") \
    for (int hc = 0; hc < 2; ++hc) { \
      _Pragma("unroll") \
      for (int nn = 0; nn < 2; ++nn) { \
        const int n = hc * 2 + nn; \
        v8f c = {}; \
        QK_TILE(c, buf, n); \
        float p[8]; \
        _Pragma("unroll") \
        for (int r = 0; r < 8; ++r) p[r] = __builtin_exp2f(c[r] - mshift); \
        float4* dst = (float4*)&attnH[(size_t)(row0 + l16) * N_CTX + \
                                      (key0) + n * 16 + 8 * half]; \
        dst[0] = make_float4(p[0], p[1], p[2], p[3]); \
        dst[1] = make_float4(p[4], p[5], p[6], p[7]); \
        _Pragma("unroll") \
        for (int r = 0; r < 8; ++r) \
          pS[wave][(nn * 16 + 8 * half + r) * PPITCH + l16] = (__bf16)p[r]; \
      } \
      __builtin_amdgcn_wave_barrier(); \
      asm volatile("s_wait_dscnt 0" ::: "memory"); \
      v16bf bp; \
      _Pragma("unroll") \
      for (int e = 0; e < 16; ++e) bp[e] = pS[wave][a_kidx(e, half) * PPITCH + l16]; \
      _Pragma("unroll") \
      for (int t = 0; t < 4; ++t) { \
        const __bf16* vp = &vT[buf][(t * 16 + l16) * KPITCH + hc * 32]; \
        v16bf av; \
        _Pragma("unroll") \
        for (int e = 0; e < 16; ++e) av[e] = vp[a_kidx(e, half)]; \
        o[t] = __builtin_amdgcn_wmma_f32_16x16x32_bf16(false, av, false, bp, (short)0, o[t], false, false); \
      } \
      __builtin_amdgcn_wave_barrier(); \
    } }

  // ================= Pass 1: row max =================
  float mrow = -INFINITY;
  LOAD_K(0); STORE_K(0);
  for (int jj = 0; jj + 2 < NJ; jj += 2) {       // steady state: branchless
    __syncthreads();
    LOAD_K((jj + 1) * KTILE);
    PREF_K(jj + 2);
    P1_BODY(0);
    STORE_K(1);
    __syncthreads();
    LOAD_K((jj + 2) * KTILE);
    PREF_K(jj + 3);
    P1_BODY(1);
    STORE_K(0);
  }
  __syncthreads();
  LOAD_K((NJ - 1) * KTILE);
  P1_BODY(0);
  STORE_K(1);
  __syncthreads();
  P1_BODY(1);
  mrow = fmaxf(mrow, __shfl_xor(mrow, 16, 32));  // merge the two key-halves

  // ================= Pass 2: row sum of exp2 =================
  float lsum = 0.f;
  LOAD_K(0); STORE_K(0);
  for (int jj = 0; jj + 2 < NJ; jj += 2) {
    __syncthreads();
    LOAD_K((jj + 1) * KTILE);
    PREF_K(jj + 2);
    P2_BODY(0);
    STORE_K(1);
    __syncthreads();
    LOAD_K((jj + 2) * KTILE);
    PREF_K(jj + 3);
    P2_BODY(1);
    STORE_K(0);
  }
  __syncthreads();
  LOAD_K((NJ - 1) * KTILE);
  P2_BODY(0);
  STORE_K(1);
  __syncthreads();
  P2_BODY(1);
  lsum += __shfl_xor(lsum, 16, 32);
  // normalizer folded into the exponent: p = exp2(c - mshift), no multiplies
  const float mshift = mrow + __builtin_log2f(lsum);

  // ============ Pass 3: emit P once, accumulate O^T = V^T.P^T ============
  v8f o[4] = {};
  LOAD_K(0); LOAD_V(0); STORE_K(0); STORE_V(0);
  for (int jj = 0; jj + 2 < NJ; jj += 2) {
    __syncthreads();
    LOAD_K((jj + 1) * KTILE); LOAD_V((jj + 1) * KTILE);
    PREF_K(jj + 2); PREF_V(jj + 2);
    P3_BODY(0, jj * KTILE);
    STORE_K(1); STORE_V(1);
    __syncthreads();
    LOAD_K((jj + 2) * KTILE); LOAD_V((jj + 2) * KTILE);
    PREF_K(jj + 3); PREF_V(jj + 3);
    P3_BODY(1, (jj + 1) * KTILE);
    STORE_K(0); STORE_V(0);
  }
  __syncthreads();
  LOAD_K((NJ - 1) * KTILE); LOAD_V((NJ - 1) * KTILE);
  P3_BODY(0, (NJ - 2) * KTILE);
  STORE_K(1); STORE_V(1);
  __syncthreads();
  P3_BODY(1, (NJ - 1) * KTILE);

  // ---- O^T C layout -> row-major out: 8 consecutive d per lane ----
  #pragma unroll
  for (int t = 0; t < 4; ++t) {
    float4* dst = (float4*)&outH[(size_t)(row0 + l16) * DHEAD + t * 16 + 8 * half];
    dst[0] = make_float4(o[t][0], o[t][1], o[t][2], o[t][3]);
    dst[1] = make_float4(o[t][4], o[t][5], o[t][6], o[t][7]);
  }
}

extern "C" void kernel_launch(void* const* d_in, const int* in_sizes, int n_in,
                              void* d_out, int out_size, void* d_ws, size_t ws_size,
                              hipStream_t stream) {
  (void)in_sizes; (void)n_in; (void)d_ws; (void)ws_size; (void)out_size;
  const float* q = (const float*)d_in[0];
  const float* k = (const float*)d_in[1];
  const float* v = (const float*)d_in[2];
  float* out  = (float*)d_out;
  float* attn = out + (size_t)BH * N_CTX * DHEAD;   // outputs concatenated flat

  dim3 grid(BH * (N_CTX / ROWS_PER_WG));            // 16 heads * 32 row-blocks = 512
  dim3 block(256);                                  // 8 wave32
  lg_attn_kernel<<<grid, block, 0, stream>>>(q, k, v, out, attn);
}
